// LinearDiffusion_28552942584321
// MI455X (gfx1250) — matrix operand order repaired and checked by reading, not compiled, not run on hardware
//
#include <hip/hip_runtime.h>

// Problem constants (reference: N=8192, D=256, E=131072, T=1, GAMMA=-1)
#define NODES 8192
#define DIM   256
#define KDIM  1024      // 4*DIM columns of Y
#define S1    0.35355339059327373f   // sqrt(1/8)
#define S3    0.61237243569579447f   // sqrt(3/8)

typedef __attribute__((ext_vector_type(16))) _Float16 v16h;
typedef __attribute__((ext_vector_type(8)))  _Float16 v8h;
typedef __attribute__((ext_vector_type(4)))  _Float16 v4h;
typedef __attribute__((ext_vector_type(8)))  float    v8f;

__device__ __forceinline__ void gatomic_add(float* p, float v) {
    unsafeAtomicAdd(p, v);   // -> global_atomic_add_f32
}

// ---------------- zero fill (k1,k2,k3 accumulators) ----------------
__global__ __launch_bounds__(256) void ld_fill_zero(float* __restrict__ p, int n) {
    int i = blockIdx.x * blockDim.x + threadIdx.x;
    if (i < n) p[i] = 0.0f;
}

// ---------------- SPMM: out[dst] += e_uv * x[src]; one wave per edge ----------------
__global__ __launch_bounds__(256) void ld_spmm(
    const float* __restrict__ x, const float* __restrict__ e,
    const int* __restrict__ src, const int* __restrict__ dst,
    float* __restrict__ out, int nE) {
    int edge = blockIdx.x * 8 + (threadIdx.x >> 5);
    if (edge >= nE) return;
    int lane = threadIdx.x & 31;
    int s = src[edge];
    int d = dst[edge];
    float ev = (s == d) ? -1.0f : e[edge];   // GAMMA self-loop override
    const float* xs = x + (size_t)s * DIM;
    float*       od = out + (size_t)d * DIM;
#pragma unroll
    for (int i = 0; i < DIM / 32; ++i) {
        int f = lane + 32 * i;               // coalesced across lanes
        gatomic_add(&od[f], ev * xs[f]);
    }
}

// ---------------- fused elementwise stages; also emit f16 panels of Y ----------------
__device__ __forceinline__ v4h cvt4(float a, float b, float c, float d, float s) {
    v4h r;
    r.x = (_Float16)(s * a); r.y = (_Float16)(s * b);
    r.z = (_Float16)(s * c); r.w = (_Float16)(s * d);
    return r;
}

// xb = x0 + k1/3 ; Y[:,0:256] = S1*x0 ; Y[:,256:512] = S3*xb
__global__ __launch_bounds__(256) void ld_stage_b(
    const float4* __restrict__ h, const float4* __restrict__ k1,
    float4* __restrict__ xb, _Float16* __restrict__ Y) {
    int i = blockIdx.x * blockDim.x + threadIdx.x;     // float4 index
    float4 x0 = h[i];
    float4 a  = k1[i];
    float4 b  = make_float4(x0.x + a.x * (1.f/3.f), x0.y + a.y * (1.f/3.f),
                            x0.z + a.z * (1.f/3.f), x0.w + a.w * (1.f/3.f));
    xb[i] = b;
    int base = i << 2;
    int row  = base >> 8;          // / DIM
    int col  = base & (DIM - 1);
    _Float16* yr = Y + (size_t)row * KDIM;
    *(v4h*)(yr + col)       = cvt4(x0.x, x0.y, x0.z, x0.w, S1);
    *(v4h*)(yr + DIM + col) = cvt4(b.x,  b.y,  b.z,  b.w,  S3);
}

// xc = x0 + k2 - k1/3 ; Y[:,512:768] = S3*xc
__global__ __launch_bounds__(256) void ld_stage_c(
    const float4* __restrict__ h, const float4* __restrict__ k1,
    const float4* __restrict__ k2, float4* __restrict__ xc, _Float16* __restrict__ Y) {
    int i = blockIdx.x * blockDim.x + threadIdx.x;
    float4 x0 = h[i], a = k1[i], b = k2[i];
    float4 c = make_float4(x0.x + b.x - a.x * (1.f/3.f), x0.y + b.y - a.y * (1.f/3.f),
                           x0.z + b.z - a.z * (1.f/3.f), x0.w + b.w - a.w * (1.f/3.f));
    xc[i] = c;
    int base = i << 2;
    int row  = base >> 8;
    int col  = base & (DIM - 1);
    _Float16* yr = Y + (size_t)row * KDIM;
    *(v4h*)(yr + 2 * DIM + col) = cvt4(c.x, c.y, c.z, c.w, S3);
}

// xd = x0 + k1 - k2 + k3 ; Y[:,768:1024] = S1*xd
__global__ __launch_bounds__(256) void ld_stage_d(
    const float4* __restrict__ h, const float4* __restrict__ k1,
    const float4* __restrict__ k2, const float4* __restrict__ k3,
    _Float16* __restrict__ Y) {
    int i = blockIdx.x * blockDim.x + threadIdx.x;
    float4 x0 = h[i], a = k1[i], b = k2[i], c = k3[i];
    float4 d = make_float4(x0.x + a.x - b.x + c.x, x0.y + a.y - b.y + c.y,
                           x0.z + a.z - b.z + c.z, x0.w + a.w - b.w + c.w);
    int base = i << 2;
    int row  = base >> 8;
    int col  = base & (DIM - 1);
    _Float16* yr = Y + (size_t)row * KDIM;
    *(v4h*)(yr + 3 * DIM + col) = cvt4(d.x, d.y, d.z, d.w, S1);
}

// ---------------- K = Y * Y^T via v_wmma_f32_16x16x32_f16 ----------------
// Block: 128 threads = 4 waves, 128x128 tile; wave: 64x64 = 4x4 WMMA accumulators.
// Both A and B fragments are rows of Y (symmetric GEMM -> no transpose loads).
__global__ __launch_bounds__(128) void ld_gram_wmma(
    const _Float16* __restrict__ Y, float* __restrict__ outK) {
    const int lane   = threadIdx.x & 31;
    const int wave   = threadIdx.x >> 5;
    const int wr     = wave >> 1;        // wave row (0..1)
    const int wc     = wave & 1;         // wave col (0..1)
    const int lanelo = lane & 15;
    const int hl     = lane >> 4;        // lane half (0 or 1)

    const int rowBase = blockIdx.y * 128 + wr * 64;
    const int colBase = blockIdx.x * 128 + wc * 64;

    v8f zero = {0.f, 0.f, 0.f, 0.f, 0.f, 0.f, 0.f, 0.f};
    v8f acc[4][4];
#pragma unroll
    for (int m = 0; m < 4; ++m)
#pragma unroll
        for (int n = 0; n < 4; ++n) acc[m][n] = zero;

    const _Float16* Arow[4];
    const _Float16* Brow[4];
#pragma unroll
    for (int m = 0; m < 4; ++m)
        Arow[m] = Y + (size_t)(rowBase + m * 16 + lanelo) * KDIM;
#pragma unroll
    for (int n = 0; n < 4; ++n)
        Brow[n] = Y + (size_t)(colBase + n * 16 + lanelo) * KDIM;

    for (int k = 0; k < KDIM; k += 32) {
        v16h a[4], b[4];
        // A 16x32 f16 layout: half 0 holds K 0..7 & 16..23, half 1 holds K 8..15 & 24..31
#pragma unroll
        for (int m = 0; m < 4; ++m) {
            const _Float16* p = Arow[m] + k + hl * 8;
            v8h c0 = *(const v8h*)(p);
            v8h c1 = *(const v8h*)(p + 16);
            a[m] = __builtin_shufflevector(c0, c1, 0, 1, 2, 3, 4, 5, 6, 7,
                                                   8, 9, 10, 11, 12, 13, 14, 15);
        }
        // B 32x16 f16 layout: half 0 holds K 0..7 & 16..23, half 1 holds K 8..15 & 24..31 (N=lane%16)
#pragma unroll
        for (int n = 0; n < 4; ++n) {
            const _Float16* q = Brow[n] + k + hl * 8;
            v8h d0 = *(const v8h*)(q);
            v8h d1 = *(const v8h*)(q + 16);
            b[n] = __builtin_shufflevector(d0, d1, 0, 1, 2, 3, 4, 5, 6, 7,
                                                   8, 9, 10, 11, 12, 13, 14, 15);
        }
#pragma unroll
        for (int m = 0; m < 4; ++m)
#pragma unroll
            for (int n = 0; n < 4; ++n)
                acc[m][n] = __builtin_amdgcn_wmma_f32_16x16x32_f16(
                    false, a[m], false, b[n], (short)0, acc[m][n], false, false);
    }

    // C/D layout: VGPR r -> M = r (lanes 0-15) / M = 8+r (lanes 16-31), N = lane%16
#pragma unroll
    for (int m = 0; m < 4; ++m) {
#pragma unroll
        for (int n = 0; n < 4; ++n) {
            int col = colBase + n * 16 + lanelo;
#pragma unroll
            for (int r = 0; r < 8; ++r) {
                int row = rowBase + m * 16 + hl * 8 + r;
                outK[(size_t)row * NODES + col] = acc[m][n][r];
            }
        }
    }
}

extern "C" void kernel_launch(void* const* d_in, const int* in_sizes, int n_in,
                              void* d_out, int out_size, void* d_ws, size_t ws_size,
                              hipStream_t stream) {
    const float* h   = (const float*)d_in[0];
    const float* e   = (const float*)d_in[1];
    const int*   src = (const int*)d_in[2];
    const int*   dst = (const int*)d_in[3];
    float* out = (float*)d_out;
    const int E = in_sizes[2];

    const size_t NB = (size_t)NODES * DIM;      // 2,097,152 floats
    float* k1 = (float*)d_ws;
    float* k2 = k1 + NB;
    float* k3 = k2 + NB;
    float* xb = k3 + NB;
    float* xc = xb + NB;
    _Float16* Y = (_Float16*)(xc + NB);          // 8192 x 1024 f16 (16 MB)

    // zero the three SPMM accumulators (contiguous)
    {
        int n = (int)(3 * NB);
        ld_fill_zero<<<(n + 255) / 256, 256, 0, stream>>>(k1, n);
    }
    int spmmBlocks = (E + 7) / 8;                 // 8 waves per 256-thread block
    int nv4 = (int)(NB / 4);
    int evBlocks = (nv4 + 255) / 256;

    // k1 = spmm(x0)
    ld_spmm<<<spmmBlocks, 256, 0, stream>>>(h, e, src, dst, k1, E);
    // xb = x0 + k1/3 ; Y panels 0,1
    ld_stage_b<<<evBlocks, 256, 0, stream>>>((const float4*)h, (const float4*)k1,
                                             (float4*)xb, Y);
    // k2 = spmm(xb)
    ld_spmm<<<spmmBlocks, 256, 0, stream>>>(xb, e, src, dst, k2, E);
    // xc = x0 + k2 - k1/3 ; Y panel 2
    ld_stage_c<<<evBlocks, 256, 0, stream>>>((const float4*)h, (const float4*)k1,
                                             (const float4*)k2, (float4*)xc, Y);
    // k3 = spmm(xc)
    ld_spmm<<<spmmBlocks, 256, 0, stream>>>(xc, e, src, dst, k3, E);
    // Y panel 3 = S1*(x0 + k1 - k2 + k3)
    ld_stage_d<<<evBlocks, 256, 0, stream>>>((const float4*)h, (const float4*)k1,
                                             (const float4*)k2, (const float4*)k3, Y);

    // K = Y * Y^T
    dim3 grid(NODES / 128, NODES / 128);   // 64 x 64 blocks
    ld_gram_wmma<<<grid, 128, 0, stream>>>(Y, out);
}
